// Face_20023137534015
// MI455X (gfx1250) — matrix operand order repaired and checked by reading, not compiled, not run on hardware
//
#include <hip/hip_runtime.h>
#include <hip/hip_bf16.h>

#define HID 32
#define VDIM 512

typedef __attribute__((ext_vector_type(16))) _Float16 v16h;
typedef __attribute__((ext_vector_type(8)))  _Float16 v8h;
typedef __attribute__((ext_vector_type(8)))  float    v8f;

// ---------------------------------------------------------------------------
// Kernel 1: moments of x over N rows -> stats[5] = {Sx0, Sx1, Sx0^2, Sx1^2, Sx0x1}
// BN over h1 = x@w1.T + b1 is derived analytically from these (h1 affine in x).
// ---------------------------------------------------------------------------
__global__ void stats_kernel(const float* __restrict__ x, int N,
                             float* __restrict__ stats) {
    int tid    = blockIdx.x * blockDim.x + threadIdx.x;
    int stride = gridDim.x * blockDim.x;
    float s0 = 0.f, s1 = 0.f, s00 = 0.f, s11 = 0.f, s01 = 0.f;
    for (int i = tid; i < N; i += stride) {
        float x0 = x[2 * i], x1 = x[2 * i + 1];
        s0 += x0; s1 += x1;
        s00 += x0 * x0; s11 += x1 * x1; s01 += x0 * x1;
    }
    #pragma unroll
    for (int off = 16; off >= 1; off >>= 1) {
        s0  += __shfl_xor(s0,  off, 32);
        s1  += __shfl_xor(s1,  off, 32);
        s00 += __shfl_xor(s00, off, 32);
        s11 += __shfl_xor(s11, off, 32);
        s01 += __shfl_xor(s01, off, 32);
    }
    if ((threadIdx.x & 31) == 0) {
        atomicAdd(&stats[0], s0);
        atomicAdd(&stats[1], s1);
        atomicAdd(&stats[2], s00);
        atomicAdd(&stats[3], s11);
        atomicAdd(&stats[4], s01);
    }
}

// ---------------------------------------------------------------------------
// Kernel 2: inverse L2 norm per visual row (one wave32 per node, b128 loads),
// computed in full f32 precision from the original data.
// ---------------------------------------------------------------------------
__global__ void invnorm_kernel(const float* __restrict__ visual, int N,
                               float* __restrict__ invn) {
    int wave = (int)(((long long)blockIdx.x * blockDim.x + threadIdx.x) >> 5);
    int lane = threadIdx.x & 31;
    if (wave >= N) return;
    const float4* row = (const float4*)(visual + (size_t)wave * VDIM);
    float ss = 0.f;
    #pragma unroll
    for (int i = 0; i < 4; ++i) {
        float4 v = row[lane + 32 * i];
        ss += v.x * v.x + v.y * v.y + v.z * v.z + v.w * v.w;
    }
    #pragma unroll
    for (int off = 16; off >= 1; off >>= 1) ss += __shfl_xor(ss, off, 32);
    if (lane == 0) invn[wave] = 1.0f / fmaxf(sqrtf(ss), 1e-8f);
}

// ---------------------------------------------------------------------------
// Kernel 2b: one-time f32 -> f16 conversion of visual. Halves the L2 gather
// traffic of the edge kernel (the roofline-dominant term: 1.9GB -> 0.96GB)
// and shrinks the gathered working set to 30.5MB (trivially L2-resident).
// ---------------------------------------------------------------------------
__global__ void vconv_kernel(const float* __restrict__ visual, long long n4,
                             _Float16* __restrict__ vh) {
    long long i = (long long)blockIdx.x * blockDim.x + threadIdx.x;
    if (i >= n4) return;
    float4 v = ((const float4*)visual)[i];
    _Float16* o = vh + 4 * i;
    o[0] = (_Float16)v.x; o[1] = (_Float16)v.y;
    o[2] = (_Float16)v.z; o[3] = (_Float16)v.w;
}

// ---------------------------------------------------------------------------
// Kernel 3: input MLP (Linear->BN->PReLU->Linear) with WMMA for layer 2.
// One wave32 per 16-node tile. Fragments per ISA f16 16x16x32 layouts:
//   A: lane holds row M=lane&15, K = (j<8?j:j+8) + 8*(lane>>4)
//   B: lane holds col N=lane&15, K = (j<8?j:j+8) + 8*(lane>>4); B[k][n] = W[n][k]
//   D: lane,vgpr r -> node r + 8*(lane>>4), feature lane&15
// ---------------------------------------------------------------------------
__global__ void mlp_kernel(const float* __restrict__ x,
                           const float* __restrict__ w1, const float* __restrict__ b1,
                           const float* __restrict__ gamma, const float* __restrict__ beta,
                           const float* __restrict__ prelu_a,
                           const float* __restrict__ w2, const float* __restrict__ b2,
                           const float* __restrict__ stats,
                           int N, float* __restrict__ h2out) {
    int waveInBlk = threadIdx.x >> 5;
    int lane      = threadIdx.x & 31;
    int tile      = blockIdx.x * (blockDim.x >> 5) + waveInBlk;
    int numTiles  = (N + 15) >> 4;
    if (tile >= numTiles) return;
    int base = tile << 4;
    int half = lane >> 4;
    int col  = lane & 15;

    // analytic BN statistics of h1 from x-moments
    float invN = 1.0f / (float)N;
    float mu0 = stats[0] * invN, mu1 = stats[1] * invN;
    float v00 = stats[2] * invN - mu0 * mu0;
    float v11 = stats[3] * invN - mu1 * mu1;
    float v01 = stats[4] * invN - mu0 * mu1;
    float slope = prelu_a[0];

    int node = base + col; if (node >= N) node = N - 1;
    float x0 = x[2 * node], x1 = x[2 * node + 1];

    v16h a;
    #pragma unroll
    for (int j = 0; j < 16; ++j) {
        int k = (j < 8 ? j : j + 8) + 8 * half;
        float wk0 = w1[2 * k], wk1 = w1[2 * k + 1];
        float h1  = wk0 * x0 + wk1 * x1 + b1[k];
        float mu  = wk0 * mu0 + wk1 * mu1 + b1[k];
        float var = wk0 * wk0 * v00 + wk1 * wk1 * v11 + 2.f * wk0 * wk1 * v01;
        float hn  = (h1 - mu) * rsqrtf(var + 1e-5f) * gamma[k] + beta[k];
        hn = (hn >= 0.f) ? hn : slope * hn;
        a[j] = (_Float16)hn;
    }

    v16h b_lo, b_hi;
    #pragma unroll
    for (int j = 0; j < 16; ++j) {
        int k = (j < 8 ? j : j + 8) + 8 * half;
        b_lo[j] = (_Float16)w2[col * HID + k];          // out feats 0..15
        b_hi[j] = (_Float16)w2[(col + 16) * HID + k];   // out feats 16..31
    }

    v8f c0 = {}; v8f c1 = {};
    c0 = __builtin_amdgcn_wmma_f32_16x16x32_f16(false, a, false, b_lo, (short)0, c0, false, false);
    c1 = __builtin_amdgcn_wmma_f32_16x16x32_f16(false, a, false, b_hi, (short)0, c1, false, false);

    float bias0 = b2[col], bias1 = b2[col + 16];
    if (base + 16 <= N) {            // wave-uniform full-tile fast path
        #pragma unroll
        for (int r = 0; r < 8; ++r) {
            int m = base + r + 8 * half;
            h2out[(size_t)m * HID + col]      = c0[r] + bias0;
            h2out[(size_t)m * HID + col + 16] = c1[r] + bias1;
        }
    } else {
        #pragma unroll
        for (int r = 0; r < 8; ++r) {
            int m = base + r + 8 * half;
            if (m < N) {
                h2out[(size_t)m * HID + col]      = c0[r] + bias0;
                h2out[(size_t)m * HID + col + 16] = c1[r] + bias1;
            }
        }
    }
}

// ---------------------------------------------------------------------------
// Kernel 4 (f16 path): edge cosine + scatter. One wave32 per edge; each lane
// gathers 2x16B of each f16 row (coalesced b128), dot accumulated in f32,
// butterfly-reduced, scaled by the f32 inverse norms. Lane f scatters
// feature f with f32 L2 atomics.
// ---------------------------------------------------------------------------
__global__ void edge_kernel_h(const int* __restrict__ ei,
                              const _Float16* __restrict__ vh,
                              const float* __restrict__ invn,
                              const float* __restrict__ h2,
                              int E, float* __restrict__ agg, float* __restrict__ deg) {
    long long wid = ((long long)blockIdx.x * blockDim.x + threadIdx.x) >> 5;
    int lane = threadIdx.x & 31;
    if (wid >= E) return;
    int src = ei[wid];
    int dst = ei[(long long)E + wid];

    const v8h* rs = (const v8h*)(vh + (size_t)src * VDIM);
    const v8h* rd = (const v8h*)(vh + (size_t)dst * VDIM);
    float dot = 0.f;
    #pragma unroll
    for (int i = 0; i < 2; ++i) {
        v8h va = rs[lane + 32 * i];
        v8h vb = rd[lane + 32 * i];
        #pragma unroll
        for (int j = 0; j < 8; ++j) dot += (float)va[j] * (float)vb[j];
    }
    #pragma unroll
    for (int off = 16; off >= 1; off >>= 1) dot += __shfl_xor(dot, off, 32);
    float w = dot * invn[src] * invn[dst];

    atomicAdd(&agg[(size_t)dst * HID + lane], w * h2[(size_t)src * HID + lane]);
    if (lane == 0) atomicAdd(&deg[dst], 1.0f);
}

// ---------------------------------------------------------------------------
// Kernel 4 (f32 fallback, used only if workspace cannot hold the f16 copy)
// ---------------------------------------------------------------------------
__global__ void edge_kernel_f(const int* __restrict__ ei,
                              const float* __restrict__ visual,
                              const float* __restrict__ invn,
                              const float* __restrict__ h2,
                              int E, float* __restrict__ agg, float* __restrict__ deg) {
    long long wid = ((long long)blockIdx.x * blockDim.x + threadIdx.x) >> 5;
    int lane = threadIdx.x & 31;
    if (wid >= E) return;
    int src = ei[wid];
    int dst = ei[(long long)E + wid];

    const float4* rs = (const float4*)(visual + (size_t)src * VDIM);
    const float4* rd = (const float4*)(visual + (size_t)dst * VDIM);
    float dot = 0.f;
    #pragma unroll
    for (int i = 0; i < 4; ++i) {
        float4 va = rs[lane + 32 * i];
        float4 vb = rd[lane + 32 * i];
        dot += va.x * vb.x + va.y * vb.y + va.z * vb.z + va.w * vb.w;
    }
    #pragma unroll
    for (int off = 16; off >= 1; off >>= 1) dot += __shfl_xor(dot, off, 32);
    float w = dot * invn[src] * invn[dst];

    atomicAdd(&agg[(size_t)dst * HID + lane], w * h2[(size_t)src * HID + lane]);
    if (lane == 0) atomicAdd(&deg[dst], 1.0f);
}

// ---------------------------------------------------------------------------
// Kernel 5: mean-normalize agg, hc = agg@wc.T + bc (WMMA), scores = hc@wp + bp.
// Same tiling/fragment layout as mlp_kernel; projection reduced across the
// 16-lane half that owns a node's 16+16 features.
// ---------------------------------------------------------------------------
__global__ void final_kernel(const float* __restrict__ agg, const float* __restrict__ deg,
                             const float* __restrict__ wc, const float* __restrict__ bc,
                             const float* __restrict__ wp, const float* __restrict__ bp,
                             int N, float* __restrict__ out) {
    int waveInBlk = threadIdx.x >> 5;
    int lane      = threadIdx.x & 31;
    int tile      = blockIdx.x * (blockDim.x >> 5) + waveInBlk;
    int numTiles  = (N + 15) >> 4;
    if (tile >= numTiles) return;
    int base = tile << 4;
    int half = lane >> 4;
    int col  = lane & 15;

    int node = base + col; if (node >= N) node = N - 1;
    float idg = 1.0f / fmaxf(deg[node], 1.0f);

    v16h a;
    #pragma unroll
    for (int j = 0; j < 16; ++j) {
        int k = (j < 8 ? j : j + 8) + 8 * half;
        a[j] = (_Float16)(agg[(size_t)node * HID + k] * idg);
    }
    v16h b_lo, b_hi;
    #pragma unroll
    for (int j = 0; j < 16; ++j) {
        int k = (j < 8 ? j : j + 8) + 8 * half;
        b_lo[j] = (_Float16)wc[col * HID + k];
        b_hi[j] = (_Float16)wc[(col + 16) * HID + k];
    }

    v8f c0 = {}; v8f c1 = {};
    c0 = __builtin_amdgcn_wmma_f32_16x16x32_f16(false, a, false, b_lo, (short)0, c0, false, false);
    c1 = __builtin_amdgcn_wmma_f32_16x16x32_f16(false, a, false, b_hi, (short)0, c1, false, false);

    float bias0 = bc[col], bias1 = bc[col + 16];
    float wp0 = wp[col], wp1 = wp[col + 16];
    float bias_p = bp[0];
    bool fullTile = (base + 16 <= N);
    #pragma unroll
    for (int r = 0; r < 8; ++r) {
        float val = (c0[r] + bias0) * wp0 + (c1[r] + bias1) * wp1;
        #pragma unroll
        for (int off = 8; off >= 1; off >>= 1) val += __shfl_xor(val, off, 16);
        if (col == 0) {
            int m = base + r + 8 * half;
            if (fullTile || m < N) out[m] = val + bias_p;
        }
    }
}

// ---------------------------------------------------------------------------
extern "C" void kernel_launch(void* const* d_in, const int* in_sizes, int n_in,
                              void* d_out, int out_size, void* d_ws, size_t ws_size,
                              hipStream_t stream) {
    const float* x       = (const float*)d_in[0];
    const float* visual  = (const float*)d_in[1];
    const int*   ei      = (const int*)d_in[2];
    const float* w1      = (const float*)d_in[3];
    const float* b1      = (const float*)d_in[4];
    const float* gamma   = (const float*)d_in[5];
    const float* beta    = (const float*)d_in[6];
    const float* prelu_a = (const float*)d_in[7];
    const float* w2      = (const float*)d_in[8];
    const float* b2      = (const float*)d_in[9];
    const float* wc      = (const float*)d_in[10];
    const float* bc      = (const float*)d_in[11];
    const float* wp      = (const float*)d_in[12];
    const float* bp      = (const float*)d_in[13];
    float* out = (float*)d_out;

    int N = in_sizes[0] / 2;
    int E = in_sizes[2] / 2;

    // workspace layout (floats): stats[8] | deg[N] | agg[32N] | h2[32N] | invn[N]
    // then (64B-aligned) vh[N*512 f16] if it fits.
    float* ws    = (float*)d_ws;
    float* stats = ws;
    float* deg   = ws + 8;
    float* aggB  = deg + N;
    float* h2    = aggB + (size_t)N * HID;
    float* invn  = h2 + (size_t)N * HID;

    size_t base_floats = 8 + (size_t)N * (2 * HID + 2);
    size_t vh_off      = (base_floats + 15) & ~(size_t)15;     // 64B aligned
    _Float16* vh       = (_Float16*)(ws + vh_off);
    size_t need_h      = vh_off * sizeof(float) + (size_t)N * VDIM * sizeof(_Float16);
    bool use_f16       = (ws_size >= need_h);

    // zero stats + deg + agg (atomic accumulation targets)
    size_t zero_bytes = (size_t)(8 + N + (size_t)N * HID) * sizeof(float);
    hipMemsetAsync(d_ws, 0, zero_bytes, stream);

    stats_kernel<<<256, 256, 0, stream>>>(x, N, stats);

    {
        long long thr = (long long)N * 32;
        int blocks = (int)((thr + 255) / 256);
        invnorm_kernel<<<blocks, 256, 0, stream>>>(visual, N, invn);
    }
    if (use_f16) {
        long long n4 = (long long)N * VDIM / 4;
        int blocks = (int)((n4 + 255) / 256);
        vconv_kernel<<<blocks, 256, 0, stream>>>(visual, n4, vh);
    }
    {
        int tiles  = (N + 15) / 16;
        int blocks = (tiles + 7) / 8;   // 8 waves per 256-thread block
        mlp_kernel<<<blocks, 256, 0, stream>>>(x, w1, b1, gamma, beta, prelu_a,
                                               w2, b2, stats, N, h2);
    }
    {
        long long thr = (long long)E * 32;
        int blocks = (int)((thr + 255) / 256);
        if (use_f16)
            edge_kernel_h<<<blocks, 256, 0, stream>>>(ei, vh, invn, h2, E, aggB, deg);
        else
            edge_kernel_f<<<blocks, 256, 0, stream>>>(ei, visual, invn, h2, E, aggB, deg);
    }
    {
        int tiles  = (N + 15) / 16;
        int blocks = (tiles + 7) / 8;
        final_kernel<<<blocks, 256, 0, stream>>>(aggB, deg, wc, bc, wp, bp, N, out);
    }
}
